// ProjEnc_76227079569804
// MI455X (gfx1250) — matrix operand order repaired and checked by reading, not compiled, not run on hardware
//
#include <hip/hip_runtime.h>
#include <hip/hip_bf16.h>
#include <math.h>

// ---------------- problem constants ----------------
#define BB   4
#define NN   2048
#define KNB  20
#define TD   64
#define GD   128
#define OBJ  200
#define IMGD 224
#define PADW 12
#define EPSV 1e-5f
#define NGRP 4

// ---------------- CDNA5 WMMA helpers (wave32, 16x16x32 bf16, f32 accum) ----------------
typedef __attribute__((ext_vector_type(16))) __bf16 v16bf;
typedef __attribute__((ext_vector_type(8)))  __bf16 v8bf;
typedef __attribute__((ext_vector_type(8)))  float  v8f;

__device__ __forceinline__ unsigned short f2bf(float f) {
  union { float f; unsigned u; } v; v.f = f;
  unsigned r = v.u + 0x7fffu + ((v.u >> 16) & 1u);   // round-to-nearest-even
  return (unsigned short)(r >> 16);
}
__device__ __forceinline__ float bf2f(unsigned short s) {
  union { unsigned u; float f; } v; v.u = ((unsigned)s) << 16; return v.f;
}

union AFrag { v16bf v; v8bf h[2]; };

// A-matrix 16x32 bf16 fragment from an LDS/global row-major [row][k] array.
// ISA layout: lane m = l%16; lanes 0-15 hold K {k0..k0+7, k0+16..k0+23},
// lanes 16-31 hold K {k0+8..k0+15, k0+24..k0+31}.
__device__ __forceinline__ v16bf load_a_frag(const unsigned short* base, int row0,
                                             int stride, int k0) {
  int l = threadIdx.x & 31;
  int off = (l < 16) ? 0 : 8;
  const __bf16* p = (const __bf16*)base + (size_t)(row0 + (l & 15)) * stride + k0 + off;
  AFrag f;
  f.h[0] = *(const v8bf*)p;
  f.h[1] = *(const v8bf*)(p + 16);
  return f.v;
}

// B-matrix 32x16 bf16 fragment; B[k][n] = W[n][k] (row-major packed weights).
// lane n = l%16; lanes 0-15 hold K {k0..k0+7, k0+16..k0+23},
// lanes 16-31 hold K {k0+8..k0+15, k0+24..k0+31}.
__device__ __forceinline__ v16bf load_b_frag(const unsigned short* w, int o0,
                                             int stride, int k0) {
  int l = threadIdx.x & 31;
  const __bf16* p = (const __bf16*)w + (size_t)(o0 + (l & 15)) * stride + k0 + ((l < 16) ? 0 : 8);
  AFrag f;
  f.h[0] = *(const v8bf*)p;
  f.h[1] = *(const v8bf*)(p + 16);
  return f.v;
}

__device__ __forceinline__ v8f wmma_bf16(v16bf a, v16bf b, v8f c) {
  // (neg_a, A, neg_b, B, c_mod, C, reuse_a, reuse_b)
  return __builtin_amdgcn_wmma_f32_16x16x32_bf16(false, a, false, b, (short)0, c, false, false);
}

#define V8F_ZERO {0.f,0.f,0.f,0.f,0.f,0.f,0.f,0.f}

// ---------------- workspace layout (bytes) ----------------
#define X_OFF     0ull
#define SQ_OFF    2097152ull
#define IDX_OFF   2129920ull
#define F_OFF     2785280ull
#define FP_OFF    4882432ull
#define SUMS_OFF  9076736ull
#define GN_OFF    9076864ull
#define BP_OFF    9076992ull
#define WG_OFF    9077120ull
#define WPJ_OFF   9109888ull
#define WIC_OFF   9142656ull
#define W1P_OFF   9175424ull
#define W2P_OFF   9470336ull
#define GRID_OFF  9765248ull
#define H1_OFF    91685248ull
#define H2_OFF    143065472ull

// ================= weight packing: fp32 -> bf16, conv weights to [o][tap*128+cin] =================
__global__ void k_pack(const float* Wg, const float* Wp, const float* Wic,
                       const float* W1, const float* W2,
                       unsigned short* wg, unsigned short* wp, unsigned short* wic,
                       unsigned short* w1p, unsigned short* w2p) {
  int i = blockIdx.x * blockDim.x + threadIdx.x;
  if (i < 16384) { wg[i] = f2bf(Wg[i]); return; }
  i -= 16384;
  if (i < 16384) { wp[i] = f2bf(Wp[i]); return; }
  i -= 16384;
  if (i < 16384) { wic[i] = f2bf(Wic[i]); return; }
  i -= 16384;
  if (i >= 2 * 147456) return;
  const float* src = W1; unsigned short* dst = w1p;
  if (i >= 147456) { i -= 147456; src = W2; dst = w2p; }
  int o = i / 1152, r = i % 1152;
  int tap = r / 128, cin = r % 128;
  int ky = tap / 3, kx = tap % 3;
  dst[o * 1152 + tap * 128 + cin] = f2bf(src[((o * 128 + cin) * 3 + ky) * 3 + kx]);
}

// ================= x = original_pc @ W_in^T + b_in ; sq = |p|^2 =================
__global__ void k_inproj(const float* opc, const float* Win, const float* bin,
                         float* x, float* sq) {
  int i = blockIdx.x * blockDim.x + threadIdx.x;
  if (i >= BB * NN * TD) return;
  int p = i / TD, c = i % TD;
  float a0 = opc[p * 3], a1 = opc[p * 3 + 1], a2 = opc[p * 3 + 2];
  x[i] = Win[c * 3] * a0 + Win[c * 3 + 1] * a1 + Win[c * 3 + 2] * a2 + bin[c];
  if (c == 0) sq[p] = a0 * a0 + a1 * a1 + a2 * a2;
}

// ================= zero fill =================
__global__ void k_zero(float* g, long long n) {
  long long i = (long long)blockIdx.x * blockDim.x + threadIdx.x;
  long long stride = (long long)gridDim.x * blockDim.x;
  for (; i < n; i += stride) g[i] = 0.f;
}

// ================= top-K nearest neighbors (per (b,n) workgroup) =================
__global__ void k_topk(const float* opc, const float* sq, int* idx) {
  __shared__ float dist[NN];
  __shared__ float rv[256];
  __shared__ int   ri[256];
  int b = blockIdx.x / NN, n = blockIdx.x % NN;
  int t = threadIdx.x;
  const float* base = opc + (size_t)b * NN * 3;
  float px = base[n * 3], py = base[n * 3 + 1], pz = base[n * 3 + 2];
  float sn = sq[b * NN + n];
  for (int j = t; j < NN; j += 256) {
    float qx = base[j * 3], qy = base[j * 3 + 1], qz = base[j * 3 + 2];
    dist[j] = sn + sq[b * NN + j] - 2.f * (px * qx + py * qy + pz * qz);
  }
  __syncthreads();
  for (int kk = 0; kk < KNB; ++kk) {
    float bv = 3.4e38f; int bi = 0x7fffffff;
    for (int j = t; j < NN; j += 256) {
      float d = dist[j];
      if (d < bv || (d == bv && j < bi)) { bv = d; bi = j; }
    }
    rv[t] = bv; ri[t] = bi;
    __syncthreads();
    for (int s = 128; s > 0; s >>= 1) {
      if (t < s) {
        if (rv[t + s] < rv[t] || (rv[t + s] == rv[t] && ri[t + s] < ri[t])) {
          rv[t] = rv[t + s]; ri[t] = ri[t + s];
        }
      }
      __syncthreads();
    }
    if (t == 0) { idx[blockIdx.x * KNB + kk] = ri[0]; dist[ri[0]] = 3.4e38f; }
    __syncthreads();
  }
}

// ================= per-batch grid parameters =================
__global__ void k_bparams(const float* pc, float* bp) {
  __shared__ float s0[256], s1[256], s2[256], s3[256];
  __shared__ float sh[4];
  int b = blockIdx.x, t = threadIdx.x;
  const float* base = pc + (size_t)b * NN * 3;
  float mnx = 3.4e38f, mny = 3.4e38f, mxx = -3.4e38f, mxy = -3.4e38f;
  for (int j = t; j < NN; j += 256) {
    float X = base[j * 3], Y = base[j * 3 + 1];
    mnx = fminf(mnx, X); mny = fminf(mny, Y);
    mxx = fmaxf(mxx, X); mxy = fmaxf(mxy, Y);
  }
  s0[t] = mnx; s1[t] = mny; s2[t] = mxx; s3[t] = mxy;
  __syncthreads();
  for (int s = 128; s > 0; s >>= 1) {
    if (t < s) {
      s0[t] = fminf(s0[t], s0[t + s]); s1[t] = fminf(s1[t], s1[t + s]);
      s2[t] = fmaxf(s2[t], s2[t + s]); s3[t] = fmaxf(s3[t], s3[t + s]);
    }
    __syncthreads();
  }
  if (t == 0) {
    sh[0] = s0[0]; sh[1] = s1[0];
    sh[2] = fmaxf(s2[0] - s0[0], s3[0] - s1[0]) / (float)(OBJ - 3);
  }
  __syncthreads();
  float pmx = sh[0], pmy = sh[1], gs = sh[2];
  mnx = 3.4e38f; mny = 3.4e38f; mxx = -3.4e38f; mxy = -3.4e38f;
  for (int j = t; j < NN; j += 256) {
    float ix = floorf((base[j * 3] - pmx) / gs);
    float iy = floorf((base[j * 3 + 1] - pmy) / gs);
    mnx = fminf(mnx, ix); mny = fminf(mny, iy);
    mxx = fmaxf(mxx, ix); mxy = fmaxf(mxy, iy);
  }
  s0[t] = mnx; s1[t] = mny; s2[t] = mxx; s3[t] = mxy;
  __syncthreads();
  for (int s = 128; s > 0; s >>= 1) {
    if (t < s) {
      s0[t] = fminf(s0[t], s0[t + s]); s1[t] = fminf(s1[t], s1[t + s]);
      s2[t] = fmaxf(s2[t], s2[t + s]); s3[t] = fmaxf(s3[t], s3[t + s]);
    }
    __syncthreads();
  }
  if (t == 0) {
    // dense spans [ixmin, ixmax+2]; center = floor((min+max)/2); shift = OBJ/2 - center - 1
    float c0 = floorf((s2[0] + 2.f + s0[0]) * 0.5f);
    float c1 = floorf((s3[0] + 2.f + s1[0]) * 0.5f);
    bp[b * 8 + 0] = pmx; bp[b * 8 + 1] = pmy; bp[b * 8 + 2] = gs;
    bp[b * 8 + 3] = (float)(OBJ / 2) - c0 - 1.f;
    bp[b * 8 + 4] = (float)(OBJ / 2) - c1 - 1.f;
  }
}

// ================= edge-conv pass 1: WMMA GEMM, accumulate GN sums only =================
__global__ void k_edge_stats(const float* x, const int* idx, const unsigned short* wg,
                             float* gsums) {
  __shared__ unsigned short feat[128 * 128];   // 128 rows x 128 feat (bf16)
  __shared__ int jrow[128], nrow[128];
  __shared__ float ls[NGRP * 2];
  int t = threadIdx.x;
  long long r0 = (long long)blockIdx.x * 128;         // flat row = ((b*N+n)*20+k)
  int b = (int)(r0 / (NN * KNB));                     // 128 | 40960 -> constant per WG
  if (t < NGRP * 2) ls[t] = 0.f;
  if (t < 128) {
    long long r = r0 + t;
    int rem = (int)(r % ((long long)NN * KNB));
    nrow[t] = rem / KNB;
    jrow[t] = idx[r];
  }
  __syncthreads();
  const float* xb = x + (size_t)b * NN * TD;
  for (int e = t; e < 128 * 128; e += 256) {
    int row = e >> 7, c = e & 127;
    float v = (c < TD) ? (xb[jrow[row] * TD + c] - xb[nrow[row] * TD + c])
                       : xb[nrow[row] * TD + (c - TD)];
    feat[e] = f2bf(v);
  }
  __syncthreads();
  int w = t >> 5;
  for (int nt = 0; nt < 8; ++nt) {
    v8f acc = V8F_ZERO;
    for (int k0 = 0; k0 < 128; k0 += 32) {
      v16bf bfr = load_b_frag(wg, nt * 16, 128, k0);
      v16bf afr = load_a_frag(feat, w * 16, 128, k0);
      acc = wmma_bf16(afr, bfr, acc);
    }
    float sA = 0.f, sB = 0.f;
#pragma unroll
    for (int i = 0; i < 8; ++i) { float v = acc[i]; sA += v; sB += v * v; }
    for (int off = 16; off > 0; off >>= 1) { sA += __shfl_xor(sA, off); sB += __shfl_xor(sB, off); }
    if ((t & 31) == 0) {
      int g = nt >> 1;                          // 16-wide N tile -> group of 32 channels
      atomicAdd(&ls[g * 2 + 0], sA);
      atomicAdd(&ls[g * 2 + 1], sB);
    }
  }
  __syncthreads();
  if (t < NGRP * 2) atomicAdd(&gsums[b * NGRP * 2 + t], ls[t]);
}

// ================= GN stats finalize =================
__global__ void k_gnstats(const float* gsums, float* gn) {
  int t = threadIdx.x;
  if (t >= BB * NGRP) return;
  float cnt = (float)NN * (float)KNB * (float)(GD / NGRP);
  float mu = gsums[t * 2] / cnt;
  float var = gsums[t * 2 + 1] / cnt - mu * mu;
  gn[t] = mu;
  gn[BB * NGRP + t] = rsqrtf(var + EPSV);
}

// ================= edge-conv pass 2: recompute y, GN + leaky + max over K -> f (bf16) =================
__global__ void k_edge_max(const float* x, const int* idx, const unsigned short* wg,
                           const float* gn, const float* gng, const float* gnb,
                           unsigned short* f) {
  __shared__ unsigned short feat[32 * 128];    // 20 valid rows, 12 zero rows
  int t = threadIdx.x;
  int p = blockIdx.x;                          // b*N + n
  int b = p / NN, n = p % NN;
  const float* xb = x + (size_t)b * NN * TD;
  for (int e = t; e < 32 * 128; e += 256) {
    int row = e >> 7, c = e & 127;
    float v = 0.f;
    if (row < KNB) {
      int j = idx[p * KNB + row];
      v = (c < TD) ? (xb[j * TD + c] - xb[n * TD + c]) : xb[n * TD + (c - TD)];
    }
    feat[e] = f2bf(v);
  }
  __syncthreads();
  int w = t >> 5, l = t & 31;
  v8f a0 = V8F_ZERO, a1 = V8F_ZERO;
  for (int k0 = 0; k0 < 128; k0 += 32) {
    v16bf bfr = load_b_frag(wg, w * 16, 128, k0);
    a0 = wmma_bf16(load_a_frag(feat, 0, 128, k0), bfr, a0);
    a1 = wmma_bf16(load_a_frag(feat, 16, 128, k0), bfr, a1);
  }
  int o = w * 16 + (l & 15);
  int g = o >> 5;
  float mu = gn[b * NGRP + g], is = gn[BB * NGRP + b * NGRP + g];
  float sc = gng[o] * is;
  float sb = gnb[o] - mu * sc;
  float m = -3.4e38f;
#pragma unroll
  for (int i = 0; i < 8; ++i) {
    float v = a0[i] * sc + sb; v = v > 0.f ? v : 0.2f * v; m = fmaxf(m, v);
  }
  int nv = (l < 16) ? 4 : 0;                   // a1 valid rows: 16..19 (lanes<16, i<4)
#pragma unroll
  for (int i = 0; i < 8; ++i) {
    if (i < nv) { float v = a1[i] * sc + sb; v = v > 0.f ? v : 0.2f * v; m = fmaxf(m, v); }
  }
  m = fmaxf(m, __shfl_xor(m, 16));
  if (l < 16) f[(size_t)p * GD + o] = f2bf(m);
}

// ================= f @ W_proj^T + b_proj (WMMA) =================
__global__ void k_proj(const unsigned short* f, const unsigned short* wproj,
                       const float* bproj, float* fp) {
  __shared__ unsigned short a_s[128 * 128];
  int t = threadIdx.x;
  long long r0 = (long long)blockIdx.x * 128;
  const unsigned* src = (const unsigned*)(f + (size_t)r0 * GD);
  unsigned* dst = (unsigned*)a_s;
  for (int e = t; e < 128 * GD / 2; e += 256) dst[e] = src[e];
  __syncthreads();
  int w = t >> 5, l = t & 31;
  for (int nt = 0; nt < 8; ++nt) {
    v8f acc = V8F_ZERO;
    for (int k0 = 0; k0 < 128; k0 += 32)
      acc = wmma_bf16(load_a_frag(a_s, w * 16, 128, k0),
                      load_b_frag(wproj, nt * 16, 128, k0), acc);
    int o = nt * 16 + (l & 15);
    float bias = bproj[o];
    int mb = w * 16 + ((l < 16) ? 0 : 8);
#pragma unroll
    for (int i = 0; i < 8; ++i)
      fp[(size_t)(r0 + mb + i) * GD + o] = acc[i] + bias;
  }
}

// ================= scatter-add into (B,200,200,128) grid =================
__global__ void k_scatter(const float* pc, const float* fp, const float* bp, float* grid) {
  int e = blockIdx.x;
  int o9 = e % 9;
  int p = e / 9;
  int b = p / NN;
  int t = threadIdx.x;                         // channel
  float px = pc[p * 3], py = pc[p * 3 + 1];
  float pmx = bp[b * 8], pmy = bp[b * 8 + 1], gs = bp[b * 8 + 2];
  float sh0 = bp[b * 8 + 3], sh1 = bp[b * 8 + 4];
  float c0 = floorf((px - pmx) / gs) + (float)(o9 / 3 - 1) + 1.f + sh0;
  float c1 = floorf((py - pmy) / gs) + (float)(o9 % 3 - 1) + 1.f + sh1;
  int i0 = (int)c0, i1 = (int)c1;
  if (c0 < 0.f || i0 >= OBJ || c1 < 0.f || i1 >= OBJ) return;
  atomicAdd(&grid[(((size_t)b * OBJ + i0) * OBJ + i1) * GD + t],
            fp[(size_t)p * GD + t]);
}

// ================= 3x3 conv implicit GEMM (WMMA bf16), fused BN (+residual) + ReLU =================
// MODE 0: input = grid f32 (implicit zero-pad to 224), epi = bn+relu       -> h1 bf16
// MODE 1: input = h1 bf16,                             epi = bn+res+relu   -> h2 bf16
template <int MODE>
__global__ void k_conv3(const void* in, const float* grid_res, const unsigned short* wp,
                        const float* g, const float* bta, const float* mvec, const float* vvec,
                        unsigned short* out) {
  __shared__ unsigned short in_s[3 * 66 * 128];    // 3 rows x 66 cols x 128ch halo tile
  int t = threadIdx.x;
  int x0 = blockIdx.x * 64;
  int y  = blockIdx.y;
  int b  = blockIdx.z;
  for (int e = t; e < 3 * 66 * 128; e += 256) {
    int rr = e / (66 * 128);
    int rem = e - rr * 66 * 128;
    int xi = rem >> 7, c = rem & 127;
    int gy = y + rr - 1, gx = x0 + xi - 1;
    unsigned short val = 0;
    if (gy >= 0 && gy < IMGD && gx >= 0 && gx < IMGD) {
      if (MODE == 0) {
        int ry = gy - PADW, rx = gx - PADW;
        if (ry >= 0 && ry < OBJ && rx >= 0 && rx < OBJ)
          val = f2bf(((const float*)in)[(((size_t)b * OBJ + ry) * OBJ + rx) * GD + c]);
      } else {
        val = ((const unsigned short*)in)[(((size_t)b * IMGD + gy) * IMGD + gx) * GD + c];
      }
    }
    in_s[e] = val;
  }
  __syncthreads();
  int w = t >> 5, l = t & 31;
  int o0 = w * 16;
  v8f acc0 = V8F_ZERO, acc1 = V8F_ZERO, acc2 = V8F_ZERO, acc3 = V8F_ZERO;
  for (int tap = 0; tap < 9; ++tap) {
    int dy = tap / 3, dx = tap % 3;
    int rowbase = dy * 66 + dx;
#pragma unroll
    for (int k0 = 0; k0 < 128; k0 += 32) {
      v16bf bfr = load_b_frag(wp, o0, 1152, tap * 128 + k0);
      acc0 = wmma_bf16(load_a_frag(in_s, rowbase + 0,  128, k0), bfr, acc0);
      acc1 = wmma_bf16(load_a_frag(in_s, rowbase + 16, 128, k0), bfr, acc1);
      acc2 = wmma_bf16(load_a_frag(in_s, rowbase + 32, 128, k0), bfr, acc2);
      acc3 = wmma_bf16(load_a_frag(in_s, rowbase + 48, 128, k0), bfr, acc3);
    }
  }
  int o = o0 + (l & 15);
  float sc = g[o] * rsqrtf(vvec[o] + EPSV);
  float sb = bta[o] - mvec[o] * sc;
#pragma unroll
  for (int mt = 0; mt < 4; ++mt) {
    v8f a = (mt == 0) ? acc0 : (mt == 1) ? acc1 : (mt == 2) ? acc2 : acc3;
#pragma unroll
    for (int i = 0; i < 8; ++i) {
      int px = x0 + mt * 16 + i + ((l < 16) ? 0 : 8);
      if (px >= IMGD) continue;
      float val = a[i] * sc + sb;
      if (MODE == 1) {
        float r = 0.f;
        int ry = y - PADW, rx = px - PADW;
        if (ry >= 0 && ry < OBJ && rx >= 0 && rx < OBJ)
          r = grid_res[(((size_t)b * OBJ + ry) * OBJ + rx) * GD + o];
        val += r;
      }
      val = fmaxf(val, 0.f);
      out[(((size_t)b * IMGD + y) * IMGD + px) * GD + o] = f2bf(val);
    }
  }
}

// ================= final: h2 @ W_imgc^T + b (WMMA), then 128->3, sigmoid, normalize =================
__global__ void k_final(const unsigned short* h2, const unsigned short* wic, const float* bic,
                        const float* Wimg, const float* bimg, float* out) {
  __shared__ unsigned short a_s[16 * 128];
  __shared__ float t_s[16 * 128];
  int t = threadIdx.x;
  long long p0 = (long long)blockIdx.x * 16;
  const unsigned* src = (const unsigned*)(h2 + (size_t)p0 * GD);
  unsigned* dst = (unsigned*)a_s;
  for (int e = t; e < 16 * GD / 2; e += 256) dst[e] = src[e];
  __syncthreads();
  int w = t >> 5, l = t & 31;
  v8f acc = V8F_ZERO;
  for (int k0 = 0; k0 < 128; k0 += 32)
    acc = wmma_bf16(load_a_frag(a_s, 0, 128, k0), load_b_frag(wic, w * 16, 128, k0), acc);
  int o = w * 16 + (l & 15);
  float bias = bic[o];
  int mb = (l < 16) ? 0 : 8;
#pragma unroll
  for (int i = 0; i < 8; ++i) t_s[(mb + i) * 128 + o] = acc[i] + bias;
  __syncthreads();
  if (t < 48) {
    int mrow = t & 15, o3 = t >> 4;
    float s = bimg[o3];
    for (int c = 0; c < GD; ++c) s += t_s[mrow * 128 + c] * Wimg[o3 * GD + c];
    float sig = 1.f / (1.f + expf(-s));
    const float meanv[3] = {0.485f, 0.456f, 0.406f};
    const float stdv[3]  = {0.229f, 0.224f, 0.225f};
    float ov = (sig - meanv[o3]) / stdv[o3];
    long long p = p0 + mrow;
    int b = (int)(p / (IMGD * IMGD));
    int rem = (int)(p % (IMGD * IMGD));
    int yy = rem / IMGD, xx = rem % IMGD;
    out[(((size_t)b * 3 + o3) * IMGD + yy) * IMGD + xx] = ov;
  }
}

// ================= host launcher =================
extern "C" void kernel_launch(void* const* d_in, const int* in_sizes, int n_in,
                              void* d_out, int out_size, void* d_ws, size_t ws_size,
                              hipStream_t stream) {
  const float* opc    = (const float*)d_in[0];
  const float* pc     = (const float*)d_in[1];
  const float* W_in   = (const float*)d_in[2];
  const float* b_in   = (const float*)d_in[3];
  const float* W_grf  = (const float*)d_in[4];
  const float* gn_g   = (const float*)d_in[5];
  const float* gn_b   = (const float*)d_in[6];
  const float* W_proj = (const float*)d_in[7];
  const float* b_proj = (const float*)d_in[8];
  const float* W1     = (const float*)d_in[9];
  const float* bn1_g  = (const float*)d_in[10];
  const float* bn1_b  = (const float*)d_in[11];
  const float* bn1_m  = (const float*)d_in[12];
  const float* bn1_v  = (const float*)d_in[13];
  const float* W2     = (const float*)d_in[14];
  const float* bn2_g  = (const float*)d_in[15];
  const float* bn2_b  = (const float*)d_in[16];
  const float* bn2_m  = (const float*)d_in[17];
  const float* bn2_v  = (const float*)d_in[18];
  const float* W_imgc = (const float*)d_in[19];
  const float* b_imgc = (const float*)d_in[20];
  const float* W_img  = (const float*)d_in[21];
  const float* b_img  = (const float*)d_in[22];

  char* ws = (char*)d_ws;
  float*          x     = (float*)(ws + X_OFF);
  float*          sq    = (float*)(ws + SQ_OFF);
  int*            idx   = (int*)(ws + IDX_OFF);
  unsigned short* fbuf  = (unsigned short*)(ws + F_OFF);
  float*          fp    = (float*)(ws + FP_OFF);
  float*          gsums = (float*)(ws + SUMS_OFF);
  float*          gn    = (float*)(ws + GN_OFF);
  float*          bp    = (float*)(ws + BP_OFF);
  unsigned short* wg    = (unsigned short*)(ws + WG_OFF);
  unsigned short* wpj   = (unsigned short*)(ws + WPJ_OFF);
  unsigned short* wic   = (unsigned short*)(ws + WIC_OFF);
  unsigned short* w1p   = (unsigned short*)(ws + W1P_OFF);
  unsigned short* w2p   = (unsigned short*)(ws + W2P_OFF);
  float*          grid  = (float*)(ws + GRID_OFF);
  unsigned short* h1    = (unsigned short*)(ws + H1_OFF);
  unsigned short* h2    = (unsigned short*)(ws + H2_OFF);
  float*          outp  = (float*)d_out;

  // 1. pack weights to bf16 (conv weights -> [o][tap*128+cin])
  k_pack<<<(3 * 16384 + 2 * 147456 + 255) / 256, 256, 0, stream>>>(
      W_grf, W_proj, W_imgc, W1, W2, wg, wpj, wic, w1p, w2p);
  // 2. input projection + squared norms
  k_inproj<<<(BB * NN * TD + 255) / 256, 256, 0, stream>>>(opc, W_in, b_in, x, sq);
  // 3. zero grid + GN sums (every call: graph replay safe)
  k_zero<<<4096, 256, 0, stream>>>(grid, (long long)BB * OBJ * OBJ * GD);
  k_zero<<<1, 256, 0, stream>>>(gsums, (long long)(BB * NGRP * 2));
  // 4. kNN top-K
  k_topk<<<BB * NN, 256, 0, stream>>>(opc, sq, idx);
  // 5. per-batch scatter parameters
  k_bparams<<<BB, 256, 0, stream>>>(pc, bp);
  // 6. edge-conv GEMM pass 1: GN statistics (WMMA)
  k_edge_stats<<<(BB * NN * KNB) / 128, 256, 0, stream>>>(x, idx, wg, gsums);
  // 7. finalize GN stats
  k_gnstats<<<1, 64, 0, stream>>>(gsums, gn);
  // 8. edge-conv GEMM pass 2 + GN + leaky + max-over-K (WMMA)
  k_edge_max<<<BB * NN, 256, 0, stream>>>(x, idx, wg, gn, gn_g, gn_b, fbuf);
  // 9. projection GEMM (WMMA)
  k_proj<<<(BB * NN) / 128, 256, 0, stream>>>(fbuf, wpj, b_proj, fp);
  // 10. scatter-add into grid (9-cell dilation, fp32 atomics)
  k_scatter<<<BB * NN * 9, GD, 0, stream>>>(pc, fp, bp, grid);
  // 11. conv1 + bn1 + relu (implicit GEMM, WMMA)
  k_conv3<0><<<dim3(4, IMGD, BB), 256, 0, stream>>>(
      (const void*)grid, grid, w1p, bn1_g, bn1_b, bn1_m, bn1_v, h1);
  // 12. conv2 + bn2 + residual + relu (implicit GEMM, WMMA)
  k_conv3<1><<<dim3(4, IMGD, BB), 256, 0, stream>>>(
      (const void*)h1, grid, w2p, bn2_g, bn2_b, bn2_m, bn2_v, h2);
  // 13. 1x1 convs + sigmoid + normalize -> NCHW output
  k_final<<<(BB * IMGD * IMGD) / 16, 256, 0, stream>>>(h2, wic, b_imgc, W_img, b_img, outp);
  (void)in_sizes; (void)n_in; (void)out_size; (void)ws_size;
}
